// SignTFAttention_54717883351257
// MI455X (gfx1250) — compile-verified, matrix operand order checked by reading, not run on hardware
//
#include <hip/hip_runtime.h>
#include <stdint.h>

typedef __attribute__((ext_vector_type(16))) __bf16        v16bf;
typedef __attribute__((ext_vector_type(8)))  float         v8f;
typedef __attribute__((ext_vector_type(4)))  unsigned int  v4u;

namespace {
constexpr int cB = 4, cT = 1024, cD = 1024, cN = 16, cH = 64;
constexpr int cBT = cB * cT;   // 4096
constexpr int cK  = cD;        // 1024 (reduction dim of dense GEMMs)

__device__ __forceinline__ unsigned short f32_to_bf16_bits(float f) {
  union { float f; unsigned int u; } a; a.f = f;
  unsigned int u = a.u;
  if ((u & 0x7fffffffu) > 0x7f800000u) return (unsigned short)((u >> 16) | 0x0040u); // quiet NaN
  return (unsigned short)((u + 0x7fffu + ((u >> 16) & 1u)) >> 16);                   // RNE
}
__device__ __forceinline__ __bf16 f32_to_bf16(float f) {
  union { unsigned short s; __bf16 b; } o; o.s = f32_to_bf16_bits(f);
  return o.b;
}

// K-index pattern for 16-bit A/B WMMA fragments (ISA 7.12.2):
// VGPR i holds K pair base: i<4 -> 2i, else 16+2(i-4); +8 for lanes 16..31.
__device__ __forceinline__ int kpat(int i, int hi) {
  return ((i < 4) ? (2 * i) : (16 + 2 * (i - 4))) + hi * 8;
}

union FragBF {
  v16bf v;
  unsigned int u[8];
};

__device__ __forceinline__ v8f v8f_zero() {
  v8f z = {0.f, 0.f, 0.f, 0.f, 0.f, 0.f, 0.f, 0.f};
  return z;
}
} // namespace

// ---------------------------------------------------------------------------
// fp32 -> bf16 conversion
// ---------------------------------------------------------------------------
__global__ void convert_f32_bf16_kernel(const float* __restrict__ src,
                                        __bf16* __restrict__ dst, int n) {
  int i = blockIdx.x * blockDim.x + threadIdx.x;
  if (i < n) dst[i] = f32_to_bf16(src[i]);
}

// ---------------------------------------------------------------------------
// bf16 WMMA GEMM:  C[m,n] = sum_k A[m,k] * W[n,k] + bias[n]
// MODE 0: fused 3-region projection (v | relu(sq) | relu(sk)), bf16 outputs
// MODE 1: single region, fp32 output (final out-projection)
// Block: 256 threads (8 waves), block tile 128x128, K-step 32.
// Each wave: 32x64 tile = 2x4 v_wmma_f32_16x16x32_bf16 accumulators.
// ---------------------------------------------------------------------------
template <int MODE>
__global__ __launch_bounds__(256)
void gemm_bf16_kernel(const __bf16* __restrict__ A,
                      const __bf16* __restrict__ W0,
                      const __bf16* __restrict__ W1,
                      const __bf16* __restrict__ W2,
                      const float* __restrict__ bias0,
                      const float* __restrict__ bias1,
                      const float* __restrict__ bias2,
                      __bf16* __restrict__ O0,
                      __bf16* __restrict__ O1,
                      __bf16* __restrict__ O2,
                      float* __restrict__ OF) {
  constexpr int LDT = 40;  // LDS row stride (bf16): 32 data + 8 pad, 16B aligned
  __shared__ __align__(16) __bf16 As[128 * LDT];
  __shared__ __align__(16) __bf16 Bs[128 * LDT];

  const int tid   = threadIdx.x;
  const int lane  = tid & 31;
  const int wid   = tid >> 5;
  const int lm    = lane & 15;
  const int hi    = lane >> 4;
  const int waveM = wid & 3;   // 4 waves along M (32 rows each)
  const int waveN = wid >> 2;  // 2 waves along N (64 cols each)

  const int m0     = blockIdx.y * 128;
  const int nGlob0 = blockIdx.x * 128;

  int region, nLoc0;
  const __bf16* W;
  const float* bias;
  if (MODE == 0) {
    region = nGlob0 >> 10;   // 0:v  1:sq  2:sk  (uniform per block)
    nLoc0  = nGlob0 & 1023;
    W    = (region == 0) ? W0 : (region == 1) ? W1 : W2;
    bias = (region == 0) ? bias0 : (region == 1) ? bias1 : bias2;
  } else {
    region = 0; nLoc0 = nGlob0; W = W0; bias = bias0;
  }

  v8f acc[2][4];
#pragma unroll
  for (int s = 0; s < 2; ++s)
#pragma unroll
    for (int t = 0; t < 4; ++t) acc[s][t] = v8f_zero();

  for (int k0 = 0; k0 < cK; k0 += 32) {
    __syncthreads();
    // Stage 128x32 bf16 tiles of A and W into LDS (uint4 = 8 bf16 per load).
#pragma unroll
    for (int c = 0; c < 2; ++c) {
      const int chunk = tid + c * 256;     // 0..511
      const int row   = chunk >> 2;        // 0..127
      const int kc    = (chunk & 3) * 8;   // 0,8,16,24
      const uint4 va = *(const uint4*)(A + (size_t)(m0 + row) * cK + k0 + kc);
      *(uint4*)&As[row * LDT + kc] = va;
      const uint4 vw = *(const uint4*)(W + (size_t)(nLoc0 + row) * cK + k0 + kc);
      *(uint4*)&Bs[row * LDT + kc] = vw;
      if (k0 + 32 < cK) {  // prefetch next K tile (global_prefetch_b8 path)
        __builtin_prefetch(A + (size_t)(m0 + row) * cK + k0 + 32 + kc, 0, 0);
        __builtin_prefetch(W + (size_t)(nLoc0 + row) * cK + k0 + 32 + kc, 0, 0);
      }
    }
    __syncthreads();

    FragBF af[2], wf[4];
#pragma unroll
    for (int s = 0; s < 2; ++s) {
      const int row = waveM * 32 + s * 16 + lm;
#pragma unroll
      for (int i = 0; i < 8; ++i)
        af[s].u[i] = *(const unsigned int*)&As[row * LDT + kpat(i, hi)];
    }
#pragma unroll
    for (int t = 0; t < 4; ++t) {
      const int row = waveN * 64 + t * 16 + lm;
#pragma unroll
      for (int i = 0; i < 8; ++i)
        wf[t].u[i] = *(const unsigned int*)&Bs[row * LDT + kpat(i, hi)];
    }
#pragma unroll
    for (int s = 0; s < 2; ++s)
#pragma unroll
      for (int t = 0; t < 4; ++t)
        acc[s][t] = __builtin_amdgcn_wmma_f32_16x16x32_bf16(
            false, af[s].v, false, wf[t].v, (short)0, acc[s][t], false, false);
  }

  // Epilogue: C/D layout is n = lane%16, m = r + 8*(lane/16).
#pragma unroll
  for (int s = 0; s < 2; ++s) {
#pragma unroll
    for (int t = 0; t < 4; ++t) {
      const int gn = nLoc0 + waveN * 64 + t * 16 + lm;
      const float bv = bias[gn];
#pragma unroll
      for (int r = 0; r < 8; ++r) {
        const int gm = m0 + waveM * 32 + s * 16 + r + 8 * hi;
        float val = acc[s][t][r] + bv;
        if (MODE == 0) {
          if (region != 0) val = fmaxf(val, 0.0f);  // relu for sq / sk
          __bf16* O = (region == 0) ? O0 : (region == 1) ? O1 : O2;
          O[(size_t)gm * cD + gn] = f32_to_bf16(val);
        } else {
          OF[(size_t)gm * cD + gn] = val;
        }
      }
    }
  }
}

// ---------------------------------------------------------------------------
// Fused sign-attention, 4 waves/block (64 t-rows), shared LDS staging:
//   stage sk(32x64) and v transposed vT(64x32) once per block per s-step
//   P    = sq_tile(16xH) . sk^T              (4 bf16 WMMAs per wave)
//   S    = kp[s] ? 0 : tanh(P)               -> coalesced NT b128 fp32 stores
//   ctx += S(16x32) . v(32xH)                (S relayout via per-wave LDS)
// ---------------------------------------------------------------------------
__global__ __launch_bounds__(128)
void sign_attn_kernel(const __bf16* __restrict__ sq,
                      const __bf16* __restrict__ sk,
                      const __bf16* __restrict__ v,
                      const unsigned char* __restrict__ kpm,
                      float* __restrict__ sign_out,
                      __bf16* __restrict__ ctx) {
  constexpr int SKT = 72;  // sk LDS row stride (bf16): 64 + 8 pad
  constexpr int VTT = 40;  // vT LDS row stride (bf16): 32 + 8 pad
  constexpr int SBT = 40;  // S bf16 stage row stride
  constexpr int SFT = 36;  // S f32 stage row stride (f32): 32 + 4 pad
  __shared__ __align__(16) __bf16 skS[32 * SKT];     // [s][h]
  __shared__ __align__(16) __bf16 vT[64 * VTT];      // [h][s]  (transposed)
  __shared__ __align__(16) __bf16 Ssh[4][16 * SBT];  // per-wave S (bf16)
  __shared__ __align__(16) float  Fsh[4][16 * SFT];  // per-wave S (f32)

  const int tid  = threadIdx.x;
  const int wid  = tid >> 5;
  const int lane = tid & 31;
  const int lm = lane & 15, hi = lane >> 4;
  const int head = blockIdx.y;       // b*N + n
  const int b    = head >> 4;
  const int nh   = head & 15;
  const int t0   = blockIdx.x * 64 + wid * 16;

  // sq A-fragments (16 x 64 -> two K=32 frags), reused across all s.
  FragBF aq[2];
  {
    const __bf16* p = sq + (size_t)(b * cT + t0 + lm) * cD + nh * cH;
#pragma unroll
    for (int s = 0; s < 2; ++s)
#pragma unroll
      for (int i = 0; i < 8; ++i)
        aq[s].u[i] = *(const unsigned int*)(p + s * 32 + kpat(i, hi));
  }

  v8f cacc[4];
#pragma unroll
  for (int t = 0; t < 4; ++t) cacc[t] = v8f_zero();

  for (int s0 = 0; s0 < cT; s0 += 32) {
    __syncthreads();  // previous iteration's LDS reads done
    // ---- stage sk tile [s][h] and v tile transposed [h][s] (coalesced) ----
#pragma unroll
    for (int c = 0; c < 2; ++c) {
      const int chunk = tid + c * 128;  // 0..255
      const int srow  = chunk >> 3;     // 0..31
      const int hc    = (chunk & 7) * 8;
      const size_t gro = (size_t)(b * cT + s0 + srow) * cD + nh * cH + hc;
      const uint4 dk = *(const uint4*)(sk + gro);
      *(uint4*)&skS[srow * SKT + hc] = dk;
      union { uint4 q; __bf16 e[8]; } uu;
      uu.q = *(const uint4*)(v + gro);
#pragma unroll
      for (int e = 0; e < 8; ++e) vT[(hc + e) * VTT + srow] = uu.e[e];
    }
    __syncthreads();

    // ---- P = sq . sk^T for two 16-wide s-tiles ----
    v8f P[2];
#pragma unroll
    for (int st = 0; st < 2; ++st) {
      P[st] = v8f_zero();
      const int srow = st * 16 + lm;  // lane's column s (= B-matrix n)
      FragBF kb0, kb1;
#pragma unroll
      for (int i = 0; i < 8; ++i) {
        kb0.u[i] = *(const unsigned int*)&skS[srow * SKT + kpat(i, hi)];
        kb1.u[i] = *(const unsigned int*)&skS[srow * SKT + 32 + kpat(i, hi)];
      }
      P[st] = __builtin_amdgcn_wmma_f32_16x16x32_bf16(
          false, aq[0].v, false, kb0.v, (short)0, P[st], false, false);
      P[st] = __builtin_amdgcn_wmma_f32_16x16x32_bf16(
          false, aq[1].v, false, kb1.v, (short)0, P[st], false, false);
    }

    // ---- mask + tanh; stage S as bf16 (for WMMA A) and f32 (for output) ----
#pragma unroll
    for (int st = 0; st < 2; ++st) {
      const int sc = s0 + st * 16 + lm;
      const bool masked = kpm[b * cT + sc] != 0;
#pragma unroll
      for (int r = 0; r < 8; ++r) {
        const int m = r + 8 * hi;
        const float val = masked ? 0.0f : tanhf(P[st][r]);
        Ssh[wid][m * SBT + st * 16 + lm] = f32_to_bf16(val);
        Fsh[wid][m * SFT + st * 16 + lm] = val;
      }
    }
    // Same-wave DS ops complete in order -> safe to read our own stage back.
    // Coalesced non-temporal b128 stores of the fp32 sign_weight tile.
#pragma unroll
    for (int c = 0; c < 4; ++c) {
      const int idx  = lane + 32 * c;   // 0..127
      const int row  = idx >> 3;        // 0..15
      const int col4 = (idx & 7) * 4;   // 0,4,...,28
      const v4u d = *(const v4u*)&Fsh[wid][row * SFT + col4];
      __builtin_nontemporal_store(
          d, (v4u*)(sign_out + ((size_t)head * cT + (t0 + row)) * cT + s0 + col4));
    }

    // ---- ctx += S . v ----
    FragBF sa;
#pragma unroll
    for (int i = 0; i < 8; ++i)
      sa.u[i] = *(const unsigned int*)&Ssh[wid][lm * SBT + kpat(i, hi)];
#pragma unroll
    for (int t = 0; t < 4; ++t) {
      FragBF vb;
      const int hrow = t * 16 + lm;  // lane's column h (= B-matrix n)
#pragma unroll
      for (int i = 0; i < 8; ++i)
        vb.u[i] = *(const unsigned int*)&vT[hrow * VTT + kpat(i, hi)];
      cacc[t] = __builtin_amdgcn_wmma_f32_16x16x32_bf16(
          false, sa.v, false, vb.v, (short)0, cacc[t], false, false);
    }
  }

  // ctx epilogue -> bf16, de-headed layout (row b*T+t, col nh*H + h)
#pragma unroll
  for (int t = 0; t < 4; ++t) {
#pragma unroll
    for (int r = 0; r < 8; ++r) {
      const int gm = b * cT + t0 + r + 8 * hi;
      const int gc = nh * cH + t * 16 + lm;
      ctx[(size_t)gm * cD + gc] = f32_to_bf16(cacc[t][r]);
    }
  }
}

// ---------------------------------------------------------------------------
// launch
// ---------------------------------------------------------------------------
extern "C" void kernel_launch(void* const* d_in, const int* in_sizes, int n_in,
                              void* d_out, int out_size, void* d_ws, size_t ws_size,
                              hipStream_t stream) {
  (void)in_sizes; (void)n_in; (void)out_size; (void)ws_size;

  const float* x          = (const float*)d_in[0];
  const unsigned char* kp = (const unsigned char*)d_in[1];
  const float* proj_w     = (const float*)d_in[2];
  const float* proj_b     = (const float*)d_in[3];
  const float* sign_w     = (const float*)d_in[4];
  const float* sign_b     = (const float*)d_in[5];
  const float* out_w      = (const float*)d_in[6];
  const float* out_b      = (const float*)d_in[7];

  float* out      = (float*)d_out;
  float* sign_out = out + (size_t)cBT * cD;  // outputs concatenated flat

  // workspace layout (bf16), total 48 MB
  constexpr size_t SZ_X = (size_t)cBT * cD;  // 4 Mi elems
  constexpr size_t SZ_W = (size_t)cD * cD;   // 1 Mi elems
  __bf16* xb   = (__bf16*)d_ws;
  __bf16* wv   = xb   + SZ_X;
  __bf16* wsq  = wv   + SZ_W;
  __bf16* wsk  = wsq  + SZ_W;
  __bf16* wo   = wsk  + SZ_W;
  __bf16* vb   = wo   + SZ_W;
  __bf16* sqb  = vb   + SZ_X;
  __bf16* skb  = sqb  + SZ_X;
  __bf16* ctxb = skb  + SZ_X;

  // 1) fp32 -> bf16 converts (x + the only weight slices we actually need:
  //    q/k of the kvq projection are dead code — the mask is exactly kp).
  convert_f32_bf16_kernel<<<(int)((SZ_X + 255) / 256), 256, 0, stream>>>(x, xb, (int)SZ_X);
  convert_f32_bf16_kernel<<<(int)((SZ_W + 255) / 256), 256, 0, stream>>>(proj_w + SZ_W, wv, (int)SZ_W);   // v slice: rows [D,2D)
  convert_f32_bf16_kernel<<<(int)((SZ_W + 255) / 256), 256, 0, stream>>>(sign_w, wsq, (int)SZ_W);          // sq slice
  convert_f32_bf16_kernel<<<(int)((SZ_W + 255) / 256), 256, 0, stream>>>(sign_w + SZ_W, wsk, (int)SZ_W);   // sk slice
  convert_f32_bf16_kernel<<<(int)((SZ_W + 255) / 256), 256, 0, stream>>>(out_w, wo, (int)SZ_W);

  // 2) fused projection GEMM: N = 3*1024 (v | relu(sq) | relu(sk))
  gemm_bf16_kernel<0><<<dim3(3 * cD / 128, cBT / 128), 256, 0, stream>>>(
      xb, wv, wsq, wsk,
      proj_b + cD, sign_b, sign_b + cD,
      vb, sqb, skb, nullptr);

  // 3) fused sign-attention (sign_weight fp32 output + ctx accumulation)
  sign_attn_kernel<<<dim3(cT / 64, cB * cN), 128, 0, stream>>>(
      sqb, skb, vb, kp, sign_out, ctxb);

  // 4) out = ctx @ out_w^T + out_b  (fp32 output)
  gemm_bf16_kernel<1><<<dim3(cD / 128, cBT / 128), 256, 0, stream>>>(
      ctxb, wo, nullptr, nullptr,
      out_b, nullptr, nullptr,
      nullptr, nullptr, nullptr, out);
}
